// EncNet_35433480192595
// MI455X (gfx1250) — matrix operand order repaired and checked by reading, not compiled, not run on hardware
//
#include <hip/hip_runtime.h>

// ---------------------------------------------------------------------------
// EncNet encoding layer, fused for MI455X (gfx1250, wave32, WMMA).
// Roofline: ~384 MB HBM traffic => ~16.5 us floor @ 23.3 TB/s.
// 8.6 GFLOP of GEMM done with v_wmma_f32_16x16x32_bf16 (large logit terms
// kept fp32) so matrix math stays far below the memory floor.
// ---------------------------------------------------------------------------

typedef __attribute__((ext_vector_type(16))) __bf16 v16bf;
typedef __attribute__((ext_vector_type(8)))  float  v8f;

#define B_     16
#define C_     128
#define SEQ_   16384
#define KC_    64
#define CHUNKS 16                         // blocks per batch
#define ROWS   64                         // seq rows per tile (4 waves x 16)
#define NITER  (SEQ_ / (CHUNKS * ROWS))   // 16 tiles per block

#define XSTR 132                          // bf16 stride for xs tile (pad)
#define ASTR 72                           // bf16 stride for a^T tile (pad)

__device__ __forceinline__ v8f vzero8() {
  v8f z;
  #pragma unroll
  for (int j = 0; j < 8; ++j) z[j] = 0.0f;
  return z;
}

// ---------------------------------------------------------------------------
// Kernel 1: stage x tile -> bf16 LDS, cross GEMM (WMMA), softmax (registers,
// shfl reductions), aggregation GEMM a^T @ xs (WMMA), atomic partials.
// ---------------------------------------------------------------------------
__global__ __launch_bounds__(128) void encnet_main(
    const float* __restrict__ x, const float* __restrict__ codewords,
    const float* __restrict__ smoothing,
    float* __restrict__ e_acc, float* __restrict__ asum)
{
  __shared__ __bf16 xs_lds[ROWS][XSTR];   // xs tile, bf16 (rows = s, cols = c)
  __shared__ __bf16 cw_lds[KC_][XSTR];    // codewords, bf16
  __shared__ __bf16 aT_lds[KC_][ASTR];    // a transposed: [code][s]
  __shared__ float  x2_lds[ROWS];         // fp32 ||x||^2 per row
  __shared__ float  cw2_lds[KC_];
  __shared__ float  sm_lds[KC_];

  const int tid  = threadIdx.x;
  const int lane = tid & 31;
  const int wave = tid >> 5;              // 0..3
  const int l15  = lane & 15;
  const int half = lane >> 4;             // 0/1
  const int bb    = blockIdx.x >> 4;      // CHUNKS = 16
  const int chunk = blockIdx.x & 15;

  // ---- phase 0: codewords -> LDS (bf16) + fp32 cw^2 + smoothing ----
  for (int idx = tid; idx < KC_ * C_; idx += 128) {
    int k = idx >> 7, c = idx & 127;
    cw_lds[k][c] = (__bf16)codewords[idx];
  }
  if (tid < KC_) {
    float s = 0.0f;
    #pragma unroll 4
    for (int c = 0; c < C_; ++c) { float w = codewords[tid * C_ + c]; s += w * w; }
    cw2_lds[tid] = s;
    sm_lds[tid]  = smoothing[tid];
  }
  __syncthreads();

  // per-lane loop-invariant softmax constants: N = l15 + 16*t
  float scw2[4], ssm[4];
  #pragma unroll
  for (int t = 0; t < 4; ++t) { int n = l15 + 16 * t; scw2[t] = cw2_lds[n]; ssm[t] = sm_lds[n]; }

  // persistent accumulators: wave owns codes [wave*16, wave*16+16), all 128 ch
  v8f acc2[8];
  #pragma unroll
  for (int nt = 0; nt < 8; ++nt) acc2[nt] = vzero8();
  float asum_acc[4] = {0.f, 0.f, 0.f, 0.f};

  const float* xb = x + (size_t)bb * C_ * SEQ_;

  for (int it = 0; it < NITER; ++it) {
    const int s0 = (chunk * NITER + it) * ROWS;

    __syncthreads();                      // previous tile fully consumed
    if (tid < ROWS) x2_lds[tid] = 0.0f;
    __syncthreads();

    // ---- stage xs tile: coalesced float4 loads, bf16 to LDS, fp32 x^2 ----
    {
      const int l16g = tid & 15;          // 16 lanes cover 64 s-values (float4)
      const int grp  = tid >> 4;          // 8 channel groups
      const int r    = l16g * 4;
      float q0 = 0.f, q1 = 0.f, q2 = 0.f, q3 = 0.f;
      for (int ch = grp; ch < C_; ch += 8) {
        const float4 v = *(const float4*)(xb + (size_t)ch * SEQ_ + s0 + r);
        if (it + 1 < NITER)               // global_prefetch_b8 for next tile
          __builtin_prefetch(xb + (size_t)ch * SEQ_ + s0 + ROWS + r, 0, 1);
        xs_lds[r + 0][ch] = (__bf16)v.x;  q0 += v.x * v.x;
        xs_lds[r + 1][ch] = (__bf16)v.y;  q1 += v.y * v.y;
        xs_lds[r + 2][ch] = (__bf16)v.z;  q2 += v.z * v.z;
        xs_lds[r + 3][ch] = (__bf16)v.w;  q3 += v.w * v.w;
      }
      atomicAdd(&x2_lds[r + 0], q0);
      atomicAdd(&x2_lds[r + 1], q1);
      atomicAdd(&x2_lds[r + 2], q2);
      atomicAdd(&x2_lds[r + 3], q3);
    }
    __syncthreads();

    // ---- GEMM1: cross[row][code] = xs @ cw^T for this wave's 16 rows ----
    // A (16x32 bf16) layout: lane M=l15(+0/+8 via half pattern on K):
//   elem i -> K = 16*half + i
    // B (32x16 bf16) layout: lane N=l15, elem i -> K = 16*half + i
    v8f acc1[4];
    #pragma unroll
    for (int t = 0; t < 4; ++t) acc1[t] = vzero8();
    const int rowA = wave * 16 + l15;
    #pragma unroll
    for (int kk = 0; kk < 4; ++kk) {      // c in chunks of 32
      v16bf afrag;
      const __bf16* pa = &xs_lds[rowA][kk * 32 + 16 * half];
      #pragma unroll
      for (int i = 0; i < 8; ++i) { afrag[i] = pa[i]; afrag[i + 8] = pa[i + 8]; }
      #pragma unroll
      for (int t = 0; t < 4; ++t) {       // code n-tiles
        v16bf bfrag;
        const __bf16* pb = &cw_lds[16 * t + l15][kk * 32 + 16 * half];
        #pragma unroll
        for (int i = 0; i < 16; ++i) bfrag[i] = pb[i];
        acc1[t] = __builtin_amdgcn_wmma_f32_16x16x32_bf16(
            false, afrag, false, bfrag, (short)0, acc1[t], false, false);
      }
    }

    // ---- softmax over K=64, fully in registers ----
    // C/D layout: elem v -> row M = v + 8*half (same row across a 16-lane
    // half), N = l15 + 16*t -> reduce over N = local over t + shfl_xor 1..8.
    #pragma unroll
    for (int v = 0; v < 8; ++v) {
      const int row = wave * 16 + v + 8 * half;
      const float x2v = x2_lds[row];
      float lg[4];
      float m = -1e30f;
      #pragma unroll
      for (int t = 0; t < 4; ++t) {
        lg[t] = ssm[t] * (x2v - 2.0f * acc1[t][v] + scw2[t]);
        m = fmaxf(m, lg[t]);
      }
      #pragma unroll
      for (int off = 1; off <= 8; off <<= 1) m = fmaxf(m, __shfl_xor(m, off));
      float s = 0.0f, ev[4];
      #pragma unroll
      for (int t = 0; t < 4; ++t) { ev[t] = __expf(lg[t] - m); s += ev[t]; }
      #pragma unroll
      for (int off = 1; off <= 8; off <<= 1) s += __shfl_xor(s, off);
      const float inv = 1.0f / s;
      #pragma unroll
      for (int t = 0; t < 4; ++t) {
        const float av = ev[t] * inv;
        aT_lds[l15 + 16 * t][row] = (__bf16)av;   // transposed for GEMM2 A
        asum_acc[t] += av;                        // column sums for e-correction
      }
    }
    __syncthreads();

    // ---- GEMM2: e[code][ch] += a^T @ xs (reduction over the 64 s rows) ----
    const int codeA = wave * 16 + l15;
    #pragma unroll
    for (int ks = 0; ks < 2; ++ks) {      // s in chunks of 32
      v16bf a2;
      const __bf16* pa = &aT_lds[codeA][ks * 32 + 16 * half];
      #pragma unroll
      for (int i = 0; i < 8; ++i) { a2[i] = pa[i]; a2[i + 8] = pa[i + 8]; }
      const int sb = ks * 32 + 16 * half;
      #pragma unroll
      for (int nt = 0; nt < 8; ++nt) {    // ch n-tiles
        v16bf b2;
        const int ch = nt * 16 + l15;
        #pragma unroll
        for (int i = 0; i < 16; ++i) b2[i] = xs_lds[sb + i][ch];
        acc2[nt] = __builtin_amdgcn_wmma_f32_16x16x32_bf16(
            false, a2, false, b2, (short)0, acc2[nt], false, false);
      }
    }
  }

  // ---- writeback: a column sums + e partials (unique addrs within block) ----
  #pragma unroll
  for (int t = 0; t < 4; ++t) asum_acc[t] += __shfl_xor(asum_acc[t], 16);
  if (half == 0) {
    #pragma unroll
    for (int t = 0; t < 4; ++t)
      atomicAdd(&asum[bb * KC_ + l15 + 16 * t], asum_acc[t]);
  }
  #pragma unroll
  for (int nt = 0; nt < 8; ++nt) {
    const int ch = nt * 16 + l15;
    #pragma unroll
    for (int v = 0; v < 8; ++v) {
      const int code = wave * 16 + v + 8 * half;
      atomicAdd(&e_acc[((size_t)bb * KC_ + code) * C_ + ch], acc2[nt][v]);
    }
  }
}

// ---------------------------------------------------------------------------
// Kernel 2: per-K finalize e, BN (batch stats over b,c), relu, mean over K.
// ---------------------------------------------------------------------------
__global__ __launch_bounds__(256) void encnet_bn(
    const float* __restrict__ e_acc, const float* __restrict__ asum,
    const float* __restrict__ codewords,
    const float* __restrict__ bn_w, const float* __restrict__ bn_b,
    float* __restrict__ e_norm)
{
  const int k = blockIdx.x;
  const int tid = threadIdx.x;
  float vals[8];
  float s1 = 0.0f, s2 = 0.0f;
  #pragma unroll
  for (int i = 0; i < 8; ++i) {
    const int idx = tid + 256 * i;        // idx = b*128 + c, 2048 total
    const int b = idx >> 7, c = idx & 127;
    const float e = e_acc[((size_t)b * KC_ + k) * C_ + c]
                  - asum[b * KC_ + k] * codewords[k * C_ + c];
    vals[i] = e; s1 += e; s2 += e * e;
  }
  __shared__ float r1[8], r2[8];
  __shared__ float mv[2];
  #pragma unroll
  for (int off = 16; off; off >>= 1) { s1 += __shfl_xor(s1, off); s2 += __shfl_xor(s2, off); }
  if ((tid & 31) == 0) { r1[tid >> 5] = s1; r2[tid >> 5] = s2; }
  __syncthreads();
  if (tid == 0) {
    float S1 = 0.f, S2 = 0.f;
    for (int w = 0; w < 8; ++w) { S1 += r1[w]; S2 += r2[w]; }
    const float mean = S1 * (1.0f / 2048.0f);
    const float var  = S2 * (1.0f / 2048.0f) - mean * mean;  // ddof=0
    mv[0] = mean; mv[1] = rsqrtf(var + 1e-5f);
  }
  __syncthreads();
  const float mean = mv[0], inv = mv[1];
  const float w = bn_w[k], bia = bn_b[k];
  #pragma unroll
  for (int i = 0; i < 8; ++i) {
    const int idx = tid + 256 * i;
    const float ebn = (vals[i] - mean) * inv * w + bia;
    atomicAdd(&e_norm[idx], fmaxf(ebn, 0.0f) * (1.0f / (float)KC_));
  }
}

// ---------------------------------------------------------------------------
// Kernel 3: scale = sigmoid(e_norm @ fc_w^T + fc_b)  (16 x 128, tiny)
// ---------------------------------------------------------------------------
__global__ __launch_bounds__(256) void encnet_fc(
    const float* __restrict__ e_norm, const float* __restrict__ fc_w,
    const float* __restrict__ fc_b, float* __restrict__ scale)
{
  __shared__ float en[B_ * C_];
  for (int i = threadIdx.x; i < B_ * C_; i += 256) en[i] = e_norm[i];
  __syncthreads();
  for (int i = threadIdx.x; i < B_ * C_; i += 256) {
    const int b = i >> 7, c = i & 127;
    float s = fc_b[c];
    const float* wr = fc_w + (size_t)c * C_;
    const float* er = en + b * C_;
    #pragma unroll 4
    for (int cc = 0; cc < C_; ++cc) s += er[cc] * wr[cc];
    scale[i] = 1.0f / (1.0f + __expf(-s));
  }
}

// ---------------------------------------------------------------------------
// Kernel 4: out = x * scale (broadcast over seq), float4 streaming.
// ---------------------------------------------------------------------------
__global__ __launch_bounds__(256) void encnet_scale(
    const float* __restrict__ x, const float* __restrict__ scale,
    float* __restrict__ out)
{
  const size_t i = (size_t)blockIdx.x * 256 + threadIdx.x;  // float4 index
  const float4 v = ((const float4*)x)[i];
  const float sc = scale[i >> 12];        // SEQ_/4 = 4096 float4 per (b,c)
  float4 o;
  o.x = v.x * sc; o.y = v.y * sc; o.z = v.z * sc; o.w = v.w * sc;
  ((float4*)out)[i] = o;
}

// ---------------------------------------------------------------------------
extern "C" void kernel_launch(void* const* d_in, const int* in_sizes, int n_in,
                              void* d_out, int out_size, void* d_ws, size_t ws_size,
                              hipStream_t stream) {
  (void)in_sizes; (void)n_in; (void)out_size; (void)ws_size;
  const float* x   = (const float*)d_in[0];
  const float* cw  = (const float*)d_in[1];
  const float* sm  = (const float*)d_in[2];
  const float* bnw = (const float*)d_in[3];
  const float* bnb = (const float*)d_in[4];
  const float* fcw = (const float*)d_in[5];
  const float* fcb = (const float*)d_in[6];
  float* out = (float*)d_out;

  char* ws = (char*)d_ws;
  float* e_acc  = (float*)(ws);                            // 16*64*128 f32
  float* asum   = (float*)(ws + 524288);                   // 16*64 f32
  float* e_norm = (float*)(ws + 524288 + 4096);            // 16*128 f32
  float* scale  = (float*)(ws + 524288 + 4096 + 8192);     // 16*128 f32

  // zero the atomic accumulators every call (graph-capturable memset node)
  hipMemsetAsync(d_ws, 0, 524288 + 4096 + 8192, stream);

  encnet_main <<<B_ * CHUNKS, 128, 0, stream>>>(x, cw, sm, e_acc, asum);
  encnet_bn   <<<KC_,          256, 0, stream>>>(e_acc, asum, cw, bnw, bnb, e_norm);
  encnet_fc   <<<1,            256, 0, stream>>>(e_norm, fcw, fcb, scale);
  encnet_scale<<<(B_ * C_ * (SEQ_ / 4)) / 256, 256, 0, stream>>>(x, scale, out);
}
